// GATCleaner_63170378989892
// MI455X (gfx1250) — compile-verified, matrix operand order checked by reading, not run on hardware
//
#include <hip/hip_runtime.h>
#include <hip/hip_bf16.h>
#include <math.h>

// ---------------------------------------------------------------------------
// GATv2 (3 layers) for gfx1250: bf16 WMMA GEMMs + atomic segment-softmax.
// ---------------------------------------------------------------------------

#define N_NODES   50000
#define N_EDGES   800000
#define E_TOT     (N_EDGES + N_NODES)   // self-loops appended
#define IN_CH     128
#define HID       32
#define HEADS     4
#define HC        (HEADS * HID)         // 128
#define NEG_SLOPE 0.2f

typedef __attribute__((ext_vector_type(16))) __bf16 bf16x16;
typedef __attribute__((ext_vector_type(8)))  float  f32x8;

union FragBF16 { bf16x16 v; uint4 u[2]; };

// Order-preserving float <-> uint encoding for atomicMax on floats.
__device__ __forceinline__ unsigned enc_f32(float f) {
    unsigned u = __float_as_uint(f);
    return u ^ ((unsigned)((int)u >> 31) | 0x80000000u);
}
__device__ __forceinline__ float dec_f32(unsigned u) {
    return __uint_as_float((u >> 31) ? (u ^ 0x80000000u) : ~u);
}
#define ENC_NEG_INF 0x007FFFFFu   // enc_f32(-inf)

// ---------------------------------------------------------------------------
// Utility kernels
// ---------------------------------------------------------------------------
__global__ void k_fill_f32(float* p, float v, int n) {
    int i = blockIdx.x * blockDim.x + threadIdx.x;
    if (i < n) p[i] = v;
}
__global__ void k_fill_u32(unsigned* p, unsigned v, int n) {
    int i = blockIdx.x * blockDim.x + threadIdx.x;
    if (i < n) p[i] = v;
}
// float4 -> 4x bf16 per thread (n4 = n/4)
__global__ void k_cvt_bf16x4(const float* __restrict__ s, __bf16* __restrict__ d, int n4) {
    int i = blockIdx.x * blockDim.x + threadIdx.x;
    if (i >= n4) return;
    float4 v = ((const float4*)s)[i];
    __bf16 o[4] = { (__bf16)v.x, (__bf16)v.y, (__bf16)v.z, (__bf16)v.w };
    *(uint2*)(d + (size_t)i * 4) = *(const uint2*)o;
}
// Wt[n*K + k] = W[k*N + n]   (bf16, so B fragments load contiguously along K)
__global__ void k_transpose_bf16(const float* __restrict__ W, __bf16* __restrict__ Wt,
                                 int K, int N) {
    int i = blockIdx.x * blockDim.x + threadIdx.x;
    if (i >= K * N) return;
    int k = i / N, n = i % N;
    Wt[(size_t)n * K + k] = (__bf16)W[(size_t)k * N + n];
}

// ---------------------------------------------------------------------------
// GEMM: C[M,N] = A[M,128] * B[128,N] + bias[N]
//   A  : bf16 row-major [M,128]
//   Bt : bf16 row-major [N,128]  (B transposed)
//   One wave -> 16x32 tile (two 16x16 accumulators sharing the A fragment);
//   K unrolled 128 = 8 x wmma_f32_16x16x32_bf16 per wave.
// ---------------------------------------------------------------------------
__global__ __launch_bounds__(256)
void k_gemm_bf16_wmma(const __bf16* __restrict__ A, const __bf16* __restrict__ Bt,
                      const float* __restrict__ bias, float* __restrict__ C,
                      int M, int N) {
    constexpr int K = 128;
    const int wave   = (blockIdx.x * blockDim.x + threadIdx.x) >> 5;
    const int lane   = threadIdx.x & 31;
    const int ntiles = N >> 5;                 // 32-wide column tiles
    const int tiles  = (M >> 4) * ntiles;
    if (wave >= tiles) return;                 // wave-uniform: EXEC stays all-ones
    const int tm = wave / ntiles;
    const int tn = wave % ntiles;

    const int r    = lane & 15;
    const int koff = (lane >> 4) << 3;         // 0 for lanes 0-15, 8 for 16-31

    const __bf16* __restrict__ arow  = A  + (size_t)(tm * 16 + r) * K;
    const __bf16* __restrict__ brow0 = Bt + (size_t)(tn * 32 + r) * K;
    const __bf16* __restrict__ brow1 = brow0 + (size_t)16 * K;

    f32x8 acc0 = {}, acc1 = {};
#pragma unroll
    for (int kk = 0; kk < K; kk += 32) {
        FragBF16 a, b0, b1;
        a.u[0]  = *(const uint4*)(arow  + kk + koff);        // K = kk+koff    .. +7
        a.u[1]  = *(const uint4*)(arow  + kk + koff + 16);   // K = kk+koff+16 .. +23
        b0.u[0] = *(const uint4*)(brow0 + kk + koff);
        b0.u[1] = *(const uint4*)(brow0 + kk + koff + 16);
        b1.u[0] = *(const uint4*)(brow1 + kk + koff);
        b1.u[1] = *(const uint4*)(brow1 + kk + koff + 16);
        acc0 = __builtin_amdgcn_wmma_f32_16x16x32_bf16(
                   false, a.v, false, b0.v, (short)0, acc0, false, false);
        acc1 = __builtin_amdgcn_wmma_f32_16x16x32_bf16(
                   false, a.v, false, b1.v, (short)0, acc1, false, false);
    }

    const int col0  = tn * 32 + r;
    const int col1  = col0 + 16;
    const int rbase = tm * 16 + ((lane >> 4) << 3);          // M = i (+8 for hi lanes)
    const float bc0 = bias[col0];
    const float bc1 = bias[col1];
#pragma unroll
    for (int i = 0; i < 8; ++i) {
        C[(size_t)(rbase + i) * N + col0] = acc0[i] + bc0;
        C[(size_t)(rbase + i) * N + col1] = acc1[i] + bc1;
    }
}

// ---------------------------------------------------------------------------
// Edge kernels. Edge e in [0,N_EDGES): src/dst arrays; e >= N_EDGES: self loop.
// One thread per (edge, head). H, C compile-time -> shift/mask index math,
// fully unrolled float4 gather clauses.
// ---------------------------------------------------------------------------
__device__ __forceinline__ void edge_nodes(int e, const int* __restrict__ src,
                                           const int* __restrict__ dst,
                                           int& s, int& d) {
    if (e < N_EDGES) { s = src[e]; d = dst[e]; }
    else             { s = e - N_EDGES; d = s; }
}

// logits[e,h] = sum_c leaky_relu(xl[s,h,c] + xr[d,h,c]) * att[h,c]
// + running segment max into nodemax[d,h] (order-encoded uint atomics)
template<int H, int C>
__global__ void k_edge_logits(const float* __restrict__ xl, const float* __restrict__ xr,
                              const int* __restrict__ src, const int* __restrict__ dst,
                              const float* __restrict__ att,
                              float* __restrict__ logits, unsigned* __restrict__ nodemax) {
    int idx = blockIdx.x * blockDim.x + threadIdx.x;
    if (idx >= E_TOT * H) return;
    const int e = idx / H, h = idx % H;
    int s, d; edge_nodes(e, src, dst, s, d);
    constexpr int hc = H * C;
    const float4* __restrict__ pl = (const float4*)(xl + (size_t)s * hc + h * C);
    const float4* __restrict__ pr = (const float4*)(xr + (size_t)d * hc + h * C);
    const float4* __restrict__ pa = (const float4*)(att + h * C);
    float acc = 0.f;
#pragma unroll
    for (int j = 0; j < (C >> 2); ++j) {
        float4 l = pl[j], r = pr[j], a = pa[j];
        float v0 = l.x + r.x, v1 = l.y + r.y, v2 = l.z + r.z, v3 = l.w + r.w;
        v0 = v0 > 0.f ? v0 : NEG_SLOPE * v0;
        v1 = v1 > 0.f ? v1 : NEG_SLOPE * v1;
        v2 = v2 > 0.f ? v2 : NEG_SLOPE * v2;
        v3 = v3 > 0.f ? v3 : NEG_SLOPE * v3;
        acc += v0 * a.x + v1 * a.y + v2 * a.z + v3 * a.w;
    }
    logits[idx] = acc;
    atomicMax(&nodemax[(size_t)d * H + h], enc_f32(acc));
}

// a = exp(logit - max[d,h]); denom[d,h] += a  (overwrites logits buffer with a)
template<int H>
__global__ void k_edge_exp(float* __restrict__ abuf,
                           const int* __restrict__ src, const int* __restrict__ dst,
                           const unsigned* __restrict__ nodemax,
                           float* __restrict__ denom) {
    int idx = blockIdx.x * blockDim.x + threadIdx.x;
    if (idx >= E_TOT * H) return;
    const int e = idx / H, h = idx % H;
    int s, d; edge_nodes(e, src, dst, s, d);
    float m = dec_f32(nodemax[(size_t)d * H + h]);
    float a = __expf(abuf[idx] - m);
    abuf[idx] = a;
    atomicAdd(&denom[(size_t)d * H + h], a);
}

// out[d,h,:] += xl[s,h,:] * (a / (denom[d,h] + 1e-16))
template<int H, int C>
__global__ void k_edge_aggr(const float* __restrict__ xl, const float* __restrict__ abuf,
                            const float* __restrict__ denom,
                            const int* __restrict__ src, const int* __restrict__ dst,
                            float* __restrict__ out) {
    int idx = blockIdx.x * blockDim.x + threadIdx.x;
    if (idx >= E_TOT * H) return;
    const int e = idx / H, h = idx % H;
    int s, d; edge_nodes(e, src, dst, s, d);
    float alpha = abuf[idx] / (denom[(size_t)d * H + h] + 1e-16f);
    constexpr int hc = H * C;
    const float* __restrict__ pl = xl  + (size_t)s * hc + h * C;
    float* __restrict__ po       = out + (size_t)d * hc + h * C;
#pragma unroll
    for (int c = 0; c < C; ++c)
        atomicAdd(&po[c], pl[c] * alpha);
}

// h = elu(out + bias)   (layers 1 & 2)
__global__ void k_elu_bias(const float* __restrict__ out, const float* __restrict__ bias,
                           float* __restrict__ h, int n, int hc) {
    int i = blockIdx.x * blockDim.x + threadIdx.x;
    if (i >= n) return;
    float v = out[i] + bias[i % hc];
    h[i] = v > 0.f ? v : (__expf(v) - 1.f);
}

// node_logits = (out + bias3) @ Wh + bh     (layer-3 epilogue + head)
__global__ void k_final_head(const float* __restrict__ out, const float* __restrict__ bias3,
                             const float* __restrict__ Wh, const float* __restrict__ bh,
                             float* __restrict__ y, int n) {
    int i = blockIdx.x * blockDim.x + threadIdx.x;
    if (i >= n) return;
    const float* __restrict__ row = out + (size_t)i * HID;
    float a0 = bh[0], a1 = bh[1];
#pragma unroll
    for (int c = 0; c < HID; ++c) {
        float t = row[c] + bias3[c];
        a0 += t * Wh[c * 2 + 0];
        a1 += t * Wh[c * 2 + 1];
    }
    y[(size_t)i * 2 + 0] = a0;
    y[(size_t)i * 2 + 1] = a1;
}

// ---------------------------------------------------------------------------
// Host side
// ---------------------------------------------------------------------------
static inline int cdiv(long long a, long long b) { return (int)((a + b - 1) / b); }

extern "C" void kernel_launch(void* const* d_in, const int* in_sizes, int n_in,
                              void* d_out, int out_size, void* d_ws, size_t ws_size,
                              hipStream_t stream) {
    const float* x   = (const float*)d_in[0];
    const int*   ei  = (const int*)d_in[1];
    const int*   src = ei;
    const int*   dst = ei + N_EDGES;

    const float* Wl1 = (const float*)d_in[2];  const float* bl1 = (const float*)d_in[3];
    const float* Wr1 = (const float*)d_in[4];  const float* br1 = (const float*)d_in[5];
    const float* at1 = (const float*)d_in[6];  const float* bi1 = (const float*)d_in[7];
    const float* Wl2 = (const float*)d_in[8];  const float* bl2 = (const float*)d_in[9];
    const float* Wr2 = (const float*)d_in[10]; const float* br2 = (const float*)d_in[11];
    const float* at2 = (const float*)d_in[12]; const float* bi2 = (const float*)d_in[13];
    const float* Wl3 = (const float*)d_in[14]; const float* bl3 = (const float*)d_in[15];
    const float* Wr3 = (const float*)d_in[16]; const float* br3 = (const float*)d_in[17];
    const float* at3 = (const float*)d_in[18]; const float* bi3 = (const float*)d_in[19];
    const float* Wh  = (const float*)d_in[20]; const float* bh  = (const float*)d_in[21];

    // ---- workspace carve (256B aligned) ----
    char* w = (char*)d_ws;
    auto carve = [&](size_t bytes) -> void* {
        void* p = w; w += (bytes + 255) & ~(size_t)255; return p;
    };
    float*    hbuf    = (float*)   carve((size_t)N_NODES * HC * 4);
    float*    xl      = (float*)   carve((size_t)N_NODES * HC * 4);
    float*    xr      = (float*)   carve((size_t)N_NODES * HC * 4);
    float*    outb    = (float*)   carve((size_t)N_NODES * HC * 4);
    float*    abuf    = (float*)   carve((size_t)E_TOT * HEADS * 4);
    unsigned* nodemax = (unsigned*)carve((size_t)N_NODES * HEADS * 4);
    float*    denom   = (float*)   carve((size_t)N_NODES * HEADS * 4);
    __bf16*   Abf     = (__bf16*)  carve((size_t)N_NODES * IN_CH * 2);
    __bf16*   Wtl     = (__bf16*)  carve((size_t)IN_CH * HC * 2);
    __bf16*   Wtr     = (__bf16*)  carve((size_t)IN_CH * HC * 2);

    const int TB = 256;
    const int nElems128 = N_NODES * HC;            // 6.4M

    // common per-layer prologue: bf16 conversions + two WMMA GEMMs
    auto run_gemms = [&](const float* xin, const float* Wl, const float* bl,
                         const float* Wr, const float* br, int Nout) {
        k_cvt_bf16x4<<<cdiv(nElems128 / 4, TB), TB, 0, stream>>>(xin, Abf, nElems128 / 4);
        k_transpose_bf16<<<cdiv((long long)IN_CH * Nout, TB), TB, 0, stream>>>(Wl, Wtl, IN_CH, Nout);
        k_transpose_bf16<<<cdiv((long long)IN_CH * Nout, TB), TB, 0, stream>>>(Wr, Wtr, IN_CH, Nout);
        const int tiles = (N_NODES / 16) * (Nout / 32);
        k_gemm_bf16_wmma<<<cdiv(tiles, 8), TB, 0, stream>>>(Abf, Wtl, bl, xl, N_NODES, Nout);
        k_gemm_bf16_wmma<<<cdiv(tiles, 8), TB, 0, stream>>>(Abf, Wtr, br, xr, N_NODES, Nout);
    };

    auto init_segments = [&](int H, int Nout) {
        k_fill_u32<<<cdiv(N_NODES * H, TB), TB, 0, stream>>>(nodemax, ENC_NEG_INF, N_NODES * H);
        k_fill_f32<<<cdiv(N_NODES * H, TB), TB, 0, stream>>>(denom, 0.f, N_NODES * H);
        k_fill_f32<<<cdiv(N_NODES * Nout, TB), TB, 0, stream>>>(outb, 0.f, N_NODES * Nout);
    };

    const int ew4 = cdiv(E_TOT * HEADS, TB);   // edge-kernel grids, H=4
    const int ew1 = cdiv(E_TOT, TB);           // H=1

    // ---- layer 1: 128 -> 4x32 (concat) + ELU ----
    run_gemms(x, Wl1, bl1, Wr1, br1, HC);
    init_segments(HEADS, HC);
    k_edge_logits<HEADS, HID><<<ew4, TB, 0, stream>>>(xl, xr, src, dst, at1, abuf, nodemax);
    k_edge_exp<HEADS>        <<<ew4, TB, 0, stream>>>(abuf, src, dst, nodemax, denom);
    k_edge_aggr<HEADS, HID>  <<<ew4, TB, 0, stream>>>(xl, abuf, denom, src, dst, outb);
    k_elu_bias<<<cdiv(nElems128, TB), TB, 0, stream>>>(outb, bi1, hbuf, nElems128, HC);

    // ---- layer 2: 128 -> 4x32 (concat) + ELU ----
    run_gemms(hbuf, Wl2, bl2, Wr2, br2, HC);
    init_segments(HEADS, HC);
    k_edge_logits<HEADS, HID><<<ew4, TB, 0, stream>>>(xl, xr, src, dst, at2, abuf, nodemax);
    k_edge_exp<HEADS>        <<<ew4, TB, 0, stream>>>(abuf, src, dst, nodemax, denom);
    k_edge_aggr<HEADS, HID>  <<<ew4, TB, 0, stream>>>(xl, abuf, denom, src, dst, outb);
    k_elu_bias<<<cdiv(nElems128, TB), TB, 0, stream>>>(outb, bi2, hbuf, nElems128, HC);

    // ---- layer 3: 128 -> 32 (1 head, no concat) ----
    run_gemms(hbuf, Wl3, bl3, Wr3, br3, HID);
    init_segments(1, HID);
    k_edge_logits<1, HID><<<ew1, TB, 0, stream>>>(xl, xr, src, dst, at3, abuf, nodemax);
    k_edge_exp<1>        <<<ew1, TB, 0, stream>>>(abuf, src, dst, nodemax, denom);
    k_edge_aggr<1, HID>  <<<ew1, TB, 0, stream>>>(xl, abuf, denom, src, dst, outb);

    // ---- classification head ----
    k_final_head<<<cdiv(N_NODES, TB), TB, 0, stream>>>(outb, bi3, Wh, bh, (float*)d_out, N_NODES);
}
